// DescriptorModuleSpeciesCat_11854109737450
// MI455X (gfx1250) — compile-verified, matrix operand order checked by reading, not run on hardware
//
#include <hip/hip_runtime.h>

// ---------------------------------------------------------------------------
// DescriptorModuleSpeciesCat for MI455X (gfx1250, wave32, WMMA)
//
// S=8, N=2048, K=64. One workgroup (128 thr = 4 wave32) per (s,n).
// - All WMMA operands load as 2x b128 per fragment (A row-major, B stored
//   transposed; the 16-bit fragment layout is two contiguous 8-half runs).
// - Weight B-fragments load straight from global (L2/WGP$-resident f16).
// - Resnet terms read from transposed LDS copies as single b128 loads.
// ---------------------------------------------------------------------------

typedef _Float16 v16h __attribute__((ext_vector_type(16)));
typedef _Float16 h8   __attribute__((ext_vector_type(8)));
typedef _Float16 h4   __attribute__((ext_vector_type(4)));
typedef float    v8f  __attribute__((ext_vector_type(8)));

#define S_DIM 8
#define N_DIM 2048
#define K_NB  64

// Load a 16x32 f16 fragment (A layout; also B layout if `base` is the
// transposed-B matrix). Works for LDS or global pointers. ldm % 8 == 0.
__device__ __forceinline__ v16h load_frag(const _Float16* base, int ldm,
                                          int row0, int k0, int lane) {
  const int m  = row0 + (lane & 15);
  const int c0 = k0 + ((lane >> 4) << 3);
  const h8 lo = *(const h8*)(base + m * ldm + c0);
  const h8 hi = *(const h8*)(base + m * ldm + c0 + 16);
  return __builtin_shufflevector(lo, hi, 0, 1, 2, 3, 4, 5, 6, 7,
                                 8, 9, 10, 11, 12, 13, 14, 15);
}

__device__ __forceinline__ v8f wmma_f16(v16h a, v16h b, v8f c) {
  return __builtin_amdgcn_wmma_f32_16x16x32_f16(false, a, false, b,
                                                (short)0, c, false, false);
}

// ------------------------- prep kernel --------------------------------------
// 1) 4-entry td lookup table (types in {0,1}): symmetrized embedding MLP
//    (2->16->32) + f MLP (32->32->31), f16 rows of 32 (slot 31 = 0).
// 2) Transposed f16 weights: w0t[64][32] = gw0^T, w1t[128][64] = gw1^T.
__global__ __launch_bounds__(256) void prep_kernel(
    const float* __restrict__ ew0, const float* __restrict__ eb0,
    const float* __restrict__ ew1, const float* __restrict__ eb1,
    const float* __restrict__ fw0, const float* __restrict__ fb0,
    const float* __restrict__ fw1, const float* __restrict__ fb1,
    const float* __restrict__ gw0, const float* __restrict__ gw1,
    _Float16* __restrict__ td4h, _Float16* __restrict__ w0t,
    _Float16* __restrict__ w1t) {
  const int tid = threadIdx.x;
  for (int i = tid; i < 32 * 64; i += 256) {
    const int r = i >> 6, c = i & 63;            // gw0[r][c], r<32 in, c<64 out
    w0t[c * 32 + r] = (_Float16)gw0[i];
  }
  for (int i = tid; i < 64 * 128; i += 256) {
    const int r = i >> 7, c = i & 127;           // gw1[r][c], r<64 in, c<128 out
    w1t[c * 64 + r] = (_Float16)gw1[i];
  }

  if (tid < 4) {
    const float a = (float)(tid >> 1);           // center type
    const float b = (float)(tid & 1);            // neighbor type
    float t[32];
    for (int j = 0; j < 32; ++j) t[j] = 0.f;
    const float pr[2][2] = {{a, b}, {b, a}};
    for (int p = 0; p < 2; ++p) {
      float h0[16];
      for (int j = 0; j < 16; ++j) {
        float v = pr[p][0] * ew0[0 * 16 + j] + pr[p][1] * ew0[1 * 16 + j] + eb0[j];
        h0[j] = v > 0.f ? v : 0.f;
      }
      for (int j = 0; j < 32; ++j) {
        float v = eb1[j];
        for (int i = 0; i < 16; ++i) v += h0[i] * ew1[i * 32 + j];
        t[j] += (v > 0.f ? v : 0.f);
      }
    }
    float f0[32];
    for (int j = 0; j < 32; ++j) {
      float v = fb0[j];
      for (int i = 0; i < 32; ++i) v += t[i] * fw0[i * 32 + j];
      f0[j] = v > 0.f ? v : 0.f;
    }
    for (int j = 0; j < 31; ++j) {
      float v = fb1[j];
      for (int i = 0; i < 32; ++i) v += f0[i] * fw1[i * 31 + j];
      td4h[tid * 32 + j] = (_Float16)(v > 0.f ? v : 0.f);
    }
    td4h[tid * 32 + 31] = (_Float16)0.f;
  }
}

// ------------------------- main kernel --------------------------------------
__global__ __launch_bounds__(128) void desc_kernel(
    const float* __restrict__ inputs, const int* __restrict__ types,
    const int* __restrict__ neigh, const float* __restrict__ length,
    const float* __restrict__ gb0, const float* __restrict__ gb1,
    const _Float16* __restrict__ td4h, const _Float16* __restrict__ w0t,
    const _Float16* __restrict__ w1t, float* __restrict__ out) {
  __shared__ __align__(16) _Float16 sX[64][32];    // s_tilde, row-major (A)
  __shared__ __align__(16) _Float16 sXt[32][64];   // s_tilde^T (resnet reads)
  __shared__ __align__(16) _Float16 sRTp[64][32];  // r_tilde K-padded (A), cols>=4 zero
  __shared__ __align__(16) _Float16 sRTt[16][64];  // r_tilde^T (A), rows>=4 zero
  __shared__ __align__(16) _Float16 sH[64][64];    // hidden, row-major (A)
  __shared__ __align__(16) _Float16 sHt[64][64];   // hidden^T (resnet reads)
  __shared__ __align__(16) _Float16 sGt[128][72];  // G^T (A for D, B^T for R)
  __shared__ __align__(16) _Float16 sRt[16][32];   // R^T (B^T), cols>=4 zero
  __shared__ __align__(16) _Float16 sR1t[16][64];  // R1^T (B^T)

  const int tid  = threadIdx.x;
  const int lane = tid & 31;
  const int wv   = tid >> 5;                       // wave id 0..3 (wave32)
  const int lg   = lane >> 4;                      // lane group 0/1
  const int blk  = blockIdx.x;
  const int s    = blk / N_DIM;
  const int n    = blk - s * N_DIM;
  const int base = s * N_DIM + n;
  const int r0   = wv * 16;
  const int m0   = r0 + (lg << 3);                 // accumulator row base

  __builtin_prefetch(w0t, 0, 3);                   // global_prefetch_b8
  __builtin_prefetch(w1t, 0, 3);

  // ---- Zero-fill padded buffers (b128 stores) ------------------------------
  {
    const uint4 z4 = {0u, 0u, 0u, 0u};
    uint4* zp = (uint4*)&sRTp[0][0];
    uint4* zt = (uint4*)&sRTt[0][0];
    uint4* zr = (uint4*)&sRt[0][0];
    for (int i = tid; i < (64 * 32 * 2) / 16; i += 128) zp[i] = z4;
    for (int i = tid; i < (16 * 64 * 2) / 16; i += 128) zt[i] = z4;
    for (int i = tid; i < (16 * 32 * 2) / 16; i += 128) zr[i] = z4;
  }
  __syncthreads();

  // ---------------- Phase A: geometry + s_tilde assembly --------------------
  if (tid < K_NB) {
    const int k  = tid;
    const int nb = neigh[base * K_NB + k];
    const bool mk = (nb < 0);
    const int idx = mk ? 0 : nb;
    const float Lx = length[0], Ly = length[1], Lz = length[2];
    const float* pi = inputs + (size_t)base * 3;
    const float* pj = inputs + ((size_t)s * N_DIM + idx) * 3;
    float dx = pj[0] - pi[0]; dx -= Lx * rintf(dx / Lx);
    float dy = pj[1] - pi[1]; dy -= Ly * rintf(dy / Ly);
    float dz = pj[2] - pi[2]; dz -= Lz * rintf(dz / Lz);
    const float rsq = dx * dx + dy * dy + dz * dz;
    const float r   = sqrtf(rsq > 0.f ? rsq : 1.f);
    const float inv = 1.f / r;
    const float u   = (r - 6.0f) / 6.0f;           // (r-R_CS)/(R_C-R_CS)
    float sw;
    if (r < 6.0f)       sw = inv;
    else if (r < 12.0f) sw = inv * (0.5f * cosf(3.14159265358979f * u) + 0.5f);
    else                sw = 0.f;
    const bool valid = (!mk) && (rsq > 0.f);
    const float sij = valid ? sw : 0.f;
    const float sc  = sij * inv;
    const _Float16 rt0 = (_Float16)sij;
    const _Float16 rt1 = (_Float16)(dx * sc);
    const _Float16 rt2 = (_Float16)(dy * sc);
    const _Float16 rt3 = (_Float16)(dz * sc);
    h4 rtv; rtv[0] = rt0; rtv[1] = rt1; rtv[2] = rt2; rtv[3] = rt3;
    *(h4*)&sRTp[k][0] = rtv;                       // one ds_store_b64
    sRTt[0][k] = rt0; sRTt[1][k] = rt1;
    sRTt[2][k] = rt2; sRTt[3][k] = rt3;

    const int ct = types[base];
    const int nt = types[s * N_DIM + idx];
    const int combo = (ct << 1) | nt;
    const h8* tdr = (const h8*)(td4h + combo * 32);
    const h8 z8 = {};
    h8 c0 = mk ? z8 : tdr[0];
    h8 c1 = mk ? z8 : tdr[1];
    h8 c2 = mk ? z8 : tdr[2];
    h8 c3 = mk ? z8 : tdr[3];
    c3[7] = (_Float16)sij;                         // s_tilde[31] = s_ij
    h8* xr = (h8*)&sX[k][0];
    xr[0] = c0; xr[1] = c1; xr[2] = c2; xr[3] = c3; // 4x ds_store_b128
#pragma unroll
    for (int j = 0; j < 8; ++j) {                  // transposed copy for resnet
      sXt[j][k]      = c0[j];
      sXt[j + 8][k]  = c1[j];
      sXt[j + 16][k] = c2[j];
      sXt[j + 24][k] = c3[j];
    }
  }
  __syncthreads();

  // ---------------- Phase B: GEMM1  H = relu(X@W0+b0) + [X,X] ---------------
  {
    const v16h a = load_frag(&sX[0][0], 32, r0, 0, lane);
#pragma unroll
    for (int c = 0; c < 4; ++c) {
      const v16h b = load_frag(w0t, 32, c * 16, 0, lane);  // global b128 x2
      v8f acc = {};
      acc = wmma_f16(a, b, acc);
      const int col  = c * 16 + (lane & 15);
      const float bias = gb0[col];
      const h8 xr = *(const h8*)&sXt[col & 31][m0];        // resnet, 1x b128
      h8 hv;
#pragma unroll
      for (int v = 0; v < 8; ++v) {
        float y = acc[v] + bias;
        y = y > 0.f ? y : 0.f;
        y += (float)xr[v];
        hv[v] = (_Float16)y;
        sH[m0 + v][col] = hv[v];                           // row-major (A)
      }
      *(h8*)&sHt[col][m0] = hv;                            // transposed, 1x b128
    }
  }
  __syncthreads();

  // ---------------- Phase C: GEMM2  G = relu(H@W1+b1) + [H,H], store G^T ----
  {
    const v16h a0 = load_frag(&sH[0][0], 64, r0, 0, lane);
    const v16h a1 = load_frag(&sH[0][0], 64, r0, 32, lane);
#pragma unroll
    for (int c = 0; c < 8; ++c) {
      v8f acc = {};
      acc = wmma_f16(a0, load_frag(w1t, 64, c * 16, 0, lane), acc);
      acc = wmma_f16(a1, load_frag(w1t, 64, c * 16, 32, lane), acc);
      const int col  = c * 16 + (lane & 15);       // feature e (0..127)
      const float bias = gb1[col];
      const h8 hr = *(const h8*)&sHt[col & 63][m0];        // resnet, 1x b128
      h8 g;
#pragma unroll
      for (int v = 0; v < 8; ++v) {
        float y = acc[v] + bias;
        y = y > 0.f ? y : 0.f;
        y += (float)hr[v];
        g[v] = (_Float16)y;
      }
      *(h8*)&sGt[col][m0] = g;                             // one ds_store_b128
    }
  }
  __syncthreads();

  // ---------------- Phase D1: R (4x16) = r_tilde^T @ G2 (wave 0) ------------
  if (wv == 0) {
    v8f acc = {};
#pragma unroll
    for (int kc = 0; kc < 2; ++kc) {
      const v16h a = load_frag(&sRTt[0][0], 64, 0, kc * 32, lane); // rows>=4 zero
      const v16h b = load_frag(&sGt[0][0], 72, 0, kc * 32, lane);  // B^T = G^T
      acc = wmma_f16(a, b, acc);
    }
    if (lane < 16) {                               // rows m=0..3 valid (lg==0)
      h4 rv;
      rv[0] = (_Float16)acc[0]; rv[1] = (_Float16)acc[1];
      rv[2] = (_Float16)acc[2]; rv[3] = (_Float16)acc[3];
      *(h4*)&sRt[lane][0] = rv;                    // R^T[n][0..3], rest zero
    }
  }
  __syncthreads();

  // ---------------- Phase D2: R1 (64x16) = r_tilde @ R ----------------------
  {
    const v16h a = load_frag(&sRTp[0][0], 32, r0, 0, lane);  // K padded 4->32
    const v16h b = load_frag(&sRt[0][0], 32, 0, 0, lane);    // B^T = R^T
    v8f acc = {};
    acc = wmma_f16(a, b, acc);
    h8 rv;
#pragma unroll
    for (int v = 0; v < 8; ++v) rv[v] = (_Float16)acc[v];
    *(h8*)&sR1t[lane & 15][m0] = rv;               // R1^T[n][m..m+7]
  }
  __syncthreads();

  // ---------------- Phase D3: D (128x16) = G^T @ R1, write out --------------
#pragma unroll
  for (int t0 = 0; t0 < 2; ++t0) {
    const int t = wv + t0 * 4;                     // e row-tile 0..7
    v8f acc = {};
#pragma unroll
    for (int kc = 0; kc < 2; ++kc) {
      const v16h a = load_frag(&sGt[0][0], 72, t * 16, kc * 32, lane);
      const v16h b = load_frag(&sR1t[0][0], 64, 0, kc * 32, lane);
      acc = wmma_f16(a, b, acc);
    }
    float* outp = out + ((size_t)blk * 128 + t * 16) * 16 + (lane & 15);
#pragma unroll
    for (int v = 0; v < 8; ++v) {
      const int e = (lg << 3) + v;
      outp[e * 16] = acc[v];
    }
  }
}

// ------------------------- launch -------------------------------------------
extern "C" void kernel_launch(void* const* d_in, const int* in_sizes, int n_in,
                              void* d_out, int out_size, void* d_ws,
                              size_t ws_size, hipStream_t stream) {
  (void)in_sizes; (void)n_in; (void)out_size; (void)ws_size;
  const float* inputs = (const float*)d_in[0];
  const int*   types  = (const int*)d_in[1];
  const int*   neigh  = (const int*)d_in[2];
  const float* length = (const float*)d_in[3];
  const float* ew0 = (const float*)d_in[4];
  const float* eb0 = (const float*)d_in[5];
  const float* ew1 = (const float*)d_in[6];
  const float* eb1 = (const float*)d_in[7];
  const float* fw0 = (const float*)d_in[8];
  const float* fb0 = (const float*)d_in[9];
  const float* fw1 = (const float*)d_in[10];
  const float* fb1 = (const float*)d_in[11];
  const float* gw0 = (const float*)d_in[12];
  const float* gb0 = (const float*)d_in[13];
  const float* gw1 = (const float*)d_in[14];
  const float* gb1 = (const float*)d_in[15];

  char* ws = (char*)d_ws;
  _Float16* td4h = (_Float16*)(ws);                  //   256 B : 4x32 td table
  _Float16* w0t  = (_Float16*)(ws + 256);            //  4096 B : gw0^T f16
  _Float16* w1t  = (_Float16*)(ws + 256 + 4096);     // 16384 B : gw1^T f16

  prep_kernel<<<1, 256, 0, stream>>>(ew0, eb0, ew1, eb1, fw0, fb0, fw1, fb1,
                                     gw0, gw1, td4h, w0t, w1t);
  desc_kernel<<<S_DIM * N_DIM, 128, 0, stream>>>(inputs, types, neigh, length,
                                                 gb0, gb1, td4h, w0t, w1t,
                                                 (float*)d_out);
}
